// PretrainGNN_5488968204774
// MI455X (gfx1250) — matrix operand order repaired and checked by reading, not compiled, not run on hardware
//
#include <hip/hip_runtime.h>

// ---------------------------------------------------------------------------
// Heterogeneous GNN (2-layer HeteroConv encoder + 4 edge-MLP decoders)
// for MI455X / gfx1250. Dense 64-dim transforms use bf16 WMMA 16x16x32
// with f32 accumulation; fragments fed by ds_load_b128 (A from row-major
// tiles, B from a pre-swizzled per-lane fragment layout in LDS).
// Irregular edge aggregation (GCN/GAT/SAGE segment ops) uses atomics.
// ---------------------------------------------------------------------------

#define NGg 20000
#define NMm 10000
#define NRr 3000
#define NBb 8000

typedef __attribute__((ext_vector_type(16))) __bf16 v16bf;
typedef __attribute__((ext_vector_type(8)))  float  v8f;
typedef __attribute__((ext_vector_type(4)))  unsigned int v4u;

struct Q2 { v4u a, b; };

__device__ __forceinline__ unsigned f2bfu(float f) {
  unsigned u = __float_as_uint(f);
  unsigned r = u + 0x7FFFu + ((u >> 16) & 1u);   // round-to-nearest-even
  return r >> 16;
}
__device__ __forceinline__ unsigned pack2bf(float lo, float hi) {
  return f2bfu(lo) | (f2bfu(hi) << 16);
}
// Build a 16-element bf16 fragment from two 16-byte LDS runs.
__device__ __forceinline__ v16bf ld_frag(const unsigned short* p0,
                                         const unsigned short* p1) {
  Q2 q;
  q.a = *(const v4u*)(const void*)p0;
  q.b = *(const v4u*)(const void*)p1;
  return __builtin_bit_cast(v16bf, q);
}
__device__ __forceinline__ float leaky02(float x) { return x > 0.f ? x : 0.2f * x; }

__device__ __forceinline__ void atomic_max_f(float* addr, float val) {
  int* ia = (int*)addr;
  int old = __float_as_int(*addr);
  while (__int_as_float(old) < val) {
    int assumed = old;
    old = atomicCAS(ia, assumed, __float_as_int(val));
    if (old == assumed) break;
  }
}

// ---------------------------------------------------------------------------
// Y[N x 64] = X[N x 64] @ W[64 x 64]   (+ optional accumulate into Y)
// bf16 WMMA 16x16x32, f32 accumulation. One wave -> 16 output rows.
// W staged in per-lane fragment order: group g = (kc*4+nb)*32+lane holds the
// 16 contiguous K values (k = kc*32 + (lane>=16?16:0) + t) for column
// n = nb*16 + (lane&15).   flags: bit0 = accumulate into existing Y
// ---------------------------------------------------------------------------
__global__ __launch_bounds__(256) void mm64_wmma(
    const float* __restrict__ X, const float* __restrict__ W,
    float* __restrict__ Y, int N, int flags) {
  __shared__ unsigned short sWf[2 * 4 * 32 * 16];   // 8 KB, fragment order
  __shared__ unsigned short sX[8][16 * 64];         // 16 KB, row-major bf16
  int tid = threadIdx.x;
  int wave = tid >> 5, lane = tid & 31;

  // ---- stage W into fragment order: one group (32B) per thread ----
  {
    int kcg = tid >> 7, nbg = (tid >> 5) & 3, lng = tid & 31;
    int n = nbg * 16 + (lng & 15);
    int kb = kcg * 32 + ((lng >> 4) ? 16 : 0);
    unsigned v[8];
#pragma unroll
    for (int t = 0; t < 8; ++t)
      v[t] = pack2bf(W[(kb + 2 * t) * 64 + n], W[(kb + 2 * t + 1) * 64 + n]);
    v4u lo = {v[0], v[1], v[2], v[3]};
    v4u hi = {v[4], v[5], v[6], v[7]};
    *(v4u*)(void*)&sWf[tid * 16]     = lo;
    *(v4u*)(void*)&sWf[tid * 16 + 8] = hi;
  }

  // ---- stage X tile (16 x 64) as bf16, float4 loads / uint2 stores ----
  int row0 = (blockIdx.x * 8 + wave) * 16;
  for (int i = lane; i < 256; i += 32) {           // 256 float4 per tile
    int r = i >> 4, c4 = i & 15;
    float4 v = {0.f, 0.f, 0.f, 0.f};
    if (row0 + r < N)
      v = ((const float4*)(X + (size_t)(row0 + r) * 64))[c4];
    uint2 p;
    p.x = pack2bf(v.x, v.y);
    p.y = pack2bf(v.z, v.w);
    *(uint2*)(void*)&sX[wave][r * 64 + c4 * 4] = p;
  }
  __syncthreads();
  if (row0 >= N) return;                           // wave-uniform exit

  v8f acc[4] = {};
  int m = lane & 15;
  int koffA = (lane >> 4) * 16;
#pragma unroll
  for (int kc = 0; kc < 2; ++kc) {
    const unsigned short* pa = &sX[wave][m * 64 + kc * 32 + koffA];
    v16bf a = ld_frag(pa, pa + 8);
#pragma unroll
    for (int nb = 0; nb < 4; ++nb) {
      const unsigned short* pb = &sWf[(((kc << 2) | nb) * 32 + lane) * 16];
      v16bf b = ld_frag(pb, pb + 8);
      acc[nb] = __builtin_amdgcn_wmma_f32_16x16x32_bf16(
          false, a, false, b, (short)0, acc[nb], false, false);
    }
  }
  // C/D layout: VGPR r, lanes 0-15 -> M=r, lanes 16-31 -> M=r+8, N=nb*16+m
  int rr = (lane >> 4) * 8;
#pragma unroll
  for (int nb = 0; nb < 4; ++nb) {
#pragma unroll
    for (int r = 0; r < 8; ++r) {
      int row = row0 + rr + r;
      if (row < N) {
        int col = nb * 16 + m;
        float v = acc[nb][r];
        if (flags & 1) v += Y[(size_t)row * 64 + col];
        Y[(size_t)row * 64 + col] = v;
      }
    }
  }
}

// ---------------------------------------------------------------------------
// Edge decoder: out[e] = relu(concat(zs[src],zd[dst]) @ W1 + b1) @ W2 + b2
// One wave -> 16 edges. K=128 via 4 bf16 WMMA chunks, 4 col tiles.
// ---------------------------------------------------------------------------
__global__ __launch_bounds__(256) void decode_wmma(
    const float* __restrict__ zs, const float* __restrict__ zd,
    const int* __restrict__ src, const int* __restrict__ dst, int E,
    const float* __restrict__ W1, const float* __restrict__ b1,
    const float* __restrict__ W2, const float* __restrict__ b2,
    float* __restrict__ out) {
  __shared__ unsigned short sW1f[4 * 4 * 32 * 16];  // 16 KB, fragment order
  __shared__ unsigned short sA[8][16 * 128];        // 32 KB, row-major bf16
  int tid = threadIdx.x;
  int wave = tid >> 5, lane = tid & 31;

  // ---- stage W1 (128x64) into fragment order: 512 groups, 2 per thread ----
  for (int g = tid; g < 512; g += 256) {
    int kcg = g >> 7, nbg = (g >> 5) & 3, lng = g & 31;
    int n = nbg * 16 + (lng & 15);
    int kb = kcg * 32 + ((lng >> 4) ? 16 : 0);
    unsigned v[8];
#pragma unroll
    for (int t = 0; t < 8; ++t)
      v[t] = pack2bf(W1[(kb + 2 * t) * 64 + n], W1[(kb + 2 * t + 1) * 64 + n]);
    v4u lo = {v[0], v[1], v[2], v[3]};
    v4u hi = {v[4], v[5], v[6], v[7]};
    *(v4u*)(void*)&sW1f[g * 16]     = lo;
    *(v4u*)(void*)&sW1f[g * 16 + 8] = hi;
  }

  // ---- gather 16 edges' concat features (16 x 128) as bf16 ----
  int e0 = (blockIdx.x * 8 + wave) * 16;
  for (int i = lane; i < 512; i += 32) {            // 512 float4 per tile
    int e = i >> 5, c4 = i & 31;                    // c4*4 = column
    float4 v = {0.f, 0.f, 0.f, 0.f};
    int ge = e0 + e;
    if (ge < E) {
      if (c4 < 16) v = ((const float4*)(zs + (size_t)src[ge] * 64))[c4];
      else         v = ((const float4*)(zd + (size_t)dst[ge] * 64))[c4 - 16];
    }
    uint2 p;
    p.x = pack2bf(v.x, v.y);
    p.y = pack2bf(v.z, v.w);
    *(uint2*)(void*)&sA[wave][e * 128 + c4 * 4] = p;
  }
  __syncthreads();
  if (e0 >= E) return;

  v8f acc[4] = {};
  int m = lane & 15;
  int koffA = (lane >> 4) * 16;
#pragma unroll
  for (int kc = 0; kc < 4; ++kc) {
    const unsigned short* pa = &sA[wave][m * 128 + kc * 32 + koffA];
    v16bf a = ld_frag(pa, pa + 8);
#pragma unroll
    for (int nb = 0; nb < 4; ++nb) {
      const unsigned short* pb = &sW1f[(((kc << 2) | nb) * 32 + lane) * 16];
      v16bf b = ld_frag(pb, pb + 8);
      acc[nb] = __builtin_amdgcn_wmma_f32_16x16x32_bf16(
          false, a, false, b, (short)0, acc[nb], false, false);
    }
  }
  // hidden = relu(acc + b1); out = hidden @ W2 + b2  (reduce over N)
  float s[8];
#pragma unroll
  for (int r = 0; r < 8; ++r) s[r] = 0.f;
#pragma unroll
  for (int nb = 0; nb < 4; ++nb) {
    float bb = b1[nb * 16 + m];
    float ww = W2[nb * 16 + m];
#pragma unroll
    for (int r = 0; r < 8; ++r)
      s[r] += fmaxf(acc[nb][r] + bb, 0.f) * ww;
  }
#pragma unroll
  for (int r = 0; r < 8; ++r) {                     // reduce within 16-lane half
    s[r] += __shfl_xor(s[r], 1, 16);
    s[r] += __shfl_xor(s[r], 2, 16);
    s[r] += __shfl_xor(s[r], 4, 16);
    s[r] += __shfl_xor(s[r], 8, 16);
  }
  if (m == 0) {
    int base = e0 + ((lane >> 4) ? 8 : 0);
    float bv = b2[0];
#pragma unroll
    for (int r = 0; r < 8; ++r) {
      int row = base + r;
      if (row < E) out[row] = s[r] + bv;
    }
  }
}

// --------------------------- small helper kernels ---------------------------

__global__ void fill_bias_k(float* __restrict__ out, const float* b0,
                            const float* b1, const float* b2, const float* b3,
                            int total) {
  int t = blockIdx.x * blockDim.x + threadIdx.x;
  if (t >= total) return;
  int j = t & 63;
  float v = 0.f;
  if (b0) v += b0[j];
  if (b1) v += b1[j];
  if (b2) v += b2[j];
  if (b3) v += b3[j];
  out[t] = v;
}

__global__ void fill_val_k(float* __restrict__ p, float v, int n) {
  int t = blockIdx.x * blockDim.x + threadIdx.x;
  if (t < n) p[t] = v;
}

__global__ void relu_k(float* __restrict__ p, int n) {
  int t = blockIdx.x * blockDim.x + threadIdx.x;
  if (t < n) p[t] = fmaxf(p[t], 0.f);
}

// ss[i] = sum_j h[i][j]*a[j]  (one wave per row)
__global__ void rowdot_k(const float* __restrict__ h, const float* __restrict__ a,
                         float* __restrict__ outv, int n) {
  int wave = (blockIdx.x * blockDim.x + threadIdx.x) >> 5;
  int lane = threadIdx.x & 31;
  if (wave >= n) return;
  float s = h[(size_t)wave * 64 + lane] * a[lane] +
            h[(size_t)wave * 64 + lane + 32] * a[lane + 32];
  s += __shfl_xor(s, 16);
  s += __shfl_xor(s, 8);
  s += __shfl_xor(s, 4);
  s += __shfl_xor(s, 2);
  s += __shfl_xor(s, 1);
  if (lane == 0) outv[wave] = s;
}

// ------------------------------ GCN kernels ---------------------------------

__global__ void count_col_k(const int* __restrict__ col, int E,
                            float* __restrict__ deg) {
  int t = blockIdx.x * blockDim.x + threadIdx.x;
  if (t < E) atomicAdd(&deg[col[t]], 1.0f);
}

__global__ void deg_to_dis_k(float* __restrict__ d, int n) {
  int t = blockIdx.x * blockDim.x + threadIdx.x;
  if (t < n) d[t] = rsqrtf(d[t] + 1.0f);   // +1 self loop; always > 0
}

__global__ void gcn_scatter_k(const int* __restrict__ row,
                              const int* __restrict__ col, int E,
                              const float* __restrict__ dis,
                              const float* __restrict__ xw,
                              float* __restrict__ out) {
  int t = blockIdx.x * blockDim.x + threadIdx.x;
  int e = t >> 5, lane = t & 31;
  if (e >= E) return;
  int r = row[e], c = col[e];
  float w = dis[r] * dis[c];
  atomicAdd(&out[(size_t)c * 64 + lane],      xw[(size_t)r * 64 + lane]      * w);
  atomicAdd(&out[(size_t)c * 64 + lane + 32], xw[(size_t)r * 64 + lane + 32] * w);
}

__global__ void gcn_self_k(const float* __restrict__ dis,
                           const float* __restrict__ xw,
                           float* __restrict__ out, int total) {
  int t = blockIdx.x * blockDim.x + threadIdx.x;
  if (t >= total) return;
  float d = dis[t >> 6];
  out[t] += xw[t] * d * d;
}

// ------------------------------ GAT kernels ---------------------------------
// edges e in [0, E+m): e >= E means PyG-style self loop on node (e - E)

__global__ void gat_max_k(const int* __restrict__ row, const int* __restrict__ col,
                          int E, int m, const float* __restrict__ ss,
                          const float* __restrict__ sd, float* __restrict__ lmax) {
  int t = blockIdx.x * blockDim.x + threadIdx.x;
  if (t >= E + m) return;
  int r, c;
  if (t < E) { r = row[t]; c = col[t]; } else { r = c = t - E; }
  atomic_max_f(&lmax[c], leaky02(ss[r] + sd[c]));
}

__global__ void gat_den_k(const int* __restrict__ row, const int* __restrict__ col,
                          int E, int m, const float* __restrict__ ss,
                          const float* __restrict__ sd,
                          const float* __restrict__ lmax, float* __restrict__ den) {
  int t = blockIdx.x * blockDim.x + threadIdx.x;
  if (t >= E + m) return;
  int r, c;
  if (t < E) { r = row[t]; c = col[t]; } else { r = c = t - E; }
  atomicAdd(&den[c], expf(leaky02(ss[r] + sd[c]) - lmax[c]));
}

__global__ void gat_scatter_k(const int* __restrict__ row, const int* __restrict__ col,
                              int E, int m, const float* __restrict__ ss,
                              const float* __restrict__ sd,
                              const float* __restrict__ lmax,
                              const float* __restrict__ den,
                              const float* __restrict__ hs, float* __restrict__ out) {
  int t = blockIdx.x * blockDim.x + threadIdx.x;
  int e = t >> 5, lane = t & 31;
  if (e >= E + m) return;
  int r, c;
  if (e < E) { r = row[e]; c = col[e]; } else { r = c = e - E; }
  float alpha = expf(leaky02(ss[r] + sd[c]) - lmax[c]) / den[c];
  atomicAdd(&out[(size_t)c * 64 + lane],      alpha * hs[(size_t)r * 64 + lane]);
  atomicAdd(&out[(size_t)c * 64 + lane + 32], alpha * hs[(size_t)r * 64 + lane + 32]);
}

// ------------------------------ SAGE kernels --------------------------------

__global__ void sage_scatter_k(const int* __restrict__ row,
                               const int* __restrict__ col, int E,
                               const float* __restrict__ xs,
                               float* __restrict__ ssum, float* __restrict__ deg) {
  int t = blockIdx.x * blockDim.x + threadIdx.x;
  int e = t >> 5, lane = t & 31;
  if (e >= E) return;
  int r = row[e], c = col[e];
  atomicAdd(&ssum[(size_t)c * 64 + lane],      xs[(size_t)r * 64 + lane]);
  atomicAdd(&ssum[(size_t)c * 64 + lane + 32], xs[(size_t)r * 64 + lane + 32]);
  if (lane == 0) atomicAdd(&deg[c], 1.0f);
}

__global__ void sage_mean_k(float* __restrict__ ssum, const float* __restrict__ deg,
                            int total) {
  int t = blockIdx.x * blockDim.x + threadIdx.x;
  if (t >= total) return;
  ssum[t] /= fmaxf(deg[t >> 6], 1.0f);
}

// ---------------------------------------------------------------------------

static inline int cdiv(int a, int b) { return (a + b - 1) / b; }

extern "C" void kernel_launch(void* const* d_in, const int* in_sizes, int n_in,
                              void* d_out, int out_size, void* d_ws, size_t ws_size,
                              hipStream_t stream) {
  (void)n_in; (void)out_size; (void)ws_size;
  const int BK = 256;
  auto F = [&](int i) { return (const float*)d_in[i]; };
  auto I = [&](int i) { return (const int*)d_in[i]; };

  // ---- data inputs ----
  const float* x_gene = F(0);
  const float* x_msig = F(1);
  const float* x_react = F(2);
  const float* x_bp = F(3);
  const int* ei_g2g = I(4);        int E_g2g = in_sizes[4] / 2;
  const int* ei_gm  = I(5);        int E_gm  = in_sizes[5] / 2;
  const int* ei_gr  = I(6);        int E_gr  = in_sizes[6] / 2;
  const int* ei_gb  = I(7);        int E_gb  = in_sizes[7] / 2;
  const int* ei_rgm = I(8);        int E_rgm = in_sizes[8] / 2;
  const int* ei_rgr = I(9);        int E_rgr = in_sizes[9] / 2;
  const int* ei_rgb = I(10);       int E_rgb = in_sizes[10] / 2;
  const int* eli_g2g = I(11);      int EL_g2g = in_sizes[11] / 2;
  const int* eli_gm  = I(12);      int EL_gm  = in_sizes[12] / 2;
  const int* eli_gr  = I(13);      int EL_gr  = in_sizes[13] / 2;
  const int* eli_gb  = I(14);      int EL_gb  = in_sizes[14] / 2;

  // ---- params (jax tree-flatten: dict keys sorted; 'dec' < 'enc') ----
  // dec: gene_bp[15..18], gene_gene[19..22], gene_msig[23..26], gene_reactome[27..30]
  //      (each: W1, W2, b1, b2)
  // enc layers at base 31 and 59; within-layer leaf offsets:
  //   g2g.W=0 b=1 | genebp.Wl=2 Wr=3 bl=4 | genemsig.Wd=5 Ws=6 a_d=7 a_s=8 b=9
  //   genereact.Wd=10..b=14 | rev_genebp.Wl=15 Wr=16 bl=17
  //   rev_genemsig.Wd=18..b=22 | rev_genereact.Wd=23..b=27
  auto LP = [&](int base, int off) { return (const float*)d_in[base + off]; };

  // ---- workspace layout (floats) ----
  float* ws = (float*)d_ws;
  float* P_hs = ws;                               // NG*64
  float* P_hd = P_hs + (size_t)NGg * 64;          // NG*64
  float* P_xw = P_hd + (size_t)NGg * 64;          // NG*64 (gcn xw / sage sum)
  float* og = P_xw + (size_t)NGg * 64;            // NG*64 (layer0 outputs)
  float* om = og + (size_t)NGg * 64;              // NM*64
  float* orr = om + (size_t)NMm * 64;             // NR*64
  float* ob = orr + (size_t)NRr * 64;             // NB*64
  float* S_ss = ob + (size_t)NBb * 64;            // NG
  float* S_sd = S_ss + NGg;                       // NG
  float* S_lmax = S_sd + NGg;                     // NG
  float* S_den = S_lmax + NGg;                    // NG
  float* S_deg = S_den + NGg;                     // NG

  // ---- output layout: (z, out) flattened ----
  float* zb = (float*)d_out;                      // z.bp      8000*64
  float* zg = zb + (size_t)NBb * 64;              // z.gene    20000*64
  float* zm = zg + (size_t)NGg * 64;              // z.msig    10000*64
  float* zr = zm + (size_t)NMm * 64;              // z.reactome 3000*64
  float* o_g2g = zr + (size_t)NRr * 64;           // sorted tuple keys
  float* o_gb = o_g2g + EL_g2g;
  float* o_gm = o_gb + EL_gb;
  float* o_gr = o_gm + EL_gm;

  auto mm = [&](const float* X, const float* W, float* Y, int N, int flags) {
    mm64_wmma<<<cdiv(N, 128), BK, 0, stream>>>(X, W, Y, N, flags);
  };
  auto fill = [&](float* p, float v, int n) {
    fill_val_k<<<cdiv(n, BK), BK, 0, stream>>>(p, v, n);
  };

  // GAT: out[col] += softmax-weighted hs[row] (+ self loops on first m nodes)
  auto run_gat = [&](const float* xs, const float* xd, const int* ei, int E,
                     int ns, int nd, int m, const float* Wd, const float* Ws,
                     const float* a_d, const float* a_s, float* out) {
    const int* row = ei; const int* col = ei + E;
    mm(xs, Ws, P_hs, ns, 0);
    mm(xd, Wd, P_hd, nd, 0);
    rowdot_k<<<cdiv(ns * 32, BK), BK, 0, stream>>>(P_hs, a_s, S_ss, ns);
    rowdot_k<<<cdiv(nd * 32, BK), BK, 0, stream>>>(P_hd, a_d, S_sd, nd);
    fill(S_lmax, -3.4e38f, nd);
    fill(S_den, 0.f, nd);
    gat_max_k<<<cdiv(E + m, BK), BK, 0, stream>>>(row, col, E, m, S_ss, S_sd, S_lmax);
    gat_den_k<<<cdiv(E + m, BK), BK, 0, stream>>>(row, col, E, m, S_ss, S_sd, S_lmax, S_den);
    gat_scatter_k<<<cdiv((E + m) * 32, BK), BK, 0, stream>>>(
        row, col, E, m, S_ss, S_sd, S_lmax, S_den, P_hs, out);
  };

  // SAGE: out += mean_agg(xs)@Wl + xd@Wr  (bl folded into out init)
  auto run_sage = [&](const float* xs, const float* xd, const int* ei, int E,
                      int nd, const float* Wl, const float* Wr, float* out) {
    const int* row = ei; const int* col = ei + E;
    fill(P_xw, 0.f, nd * 64);
    fill(S_deg, 0.f, nd);
    sage_scatter_k<<<cdiv(E * 32, BK), BK, 0, stream>>>(row, col, E, xs, P_xw, S_deg);
    sage_mean_k<<<cdiv(nd * 64, BK), BK, 0, stream>>>(P_xw, S_deg, nd * 64);
    mm(P_xw, Wl, out, nd, 1);
    mm(xd, Wr, out, nd, 1);
  };

  // ---- one HeteroConv layer (sum aggregation + relu) ----
  auto run_layer = [&](int pb, const float* xg, const float* xm, const float* xr,
                       const float* xb2, float* yg, float* ym, float* yr, float* yb) {
    // gene = gcn(g2g) + gat(rev_genemsig) + gat(rev_genereact) + sage(rev_genebp)
    fill_bias_k<<<cdiv(NGg * 64, BK), BK, 0, stream>>>(
        yg, LP(pb, 1), LP(pb, 22), LP(pb, 27), LP(pb, 17), NGg * 64);
    // GCN
    mm(xg, LP(pb, 0), P_xw, NGg, 0);
    fill(S_deg, 0.f, NGg);
    count_col_k<<<cdiv(E_g2g, BK), BK, 0, stream>>>(ei_g2g + E_g2g, E_g2g, S_deg);
    deg_to_dis_k<<<cdiv(NGg, BK), BK, 0, stream>>>(S_deg, NGg);
    gcn_scatter_k<<<cdiv(E_g2g * 32, BK), BK, 0, stream>>>(
        ei_g2g, ei_g2g + E_g2g, E_g2g, S_deg, P_xw, yg);
    gcn_self_k<<<cdiv(NGg * 64, BK), BK, 0, stream>>>(S_deg, P_xw, yg, NGg * 64);
    // GATs into gene
    run_gat(xm, xg, ei_rgm, E_rgm, NMm, NGg, NMm,
            LP(pb, 18), LP(pb, 19), LP(pb, 20), LP(pb, 21), yg);
    run_gat(xr, xg, ei_rgr, E_rgr, NRr, NGg, NRr,
            LP(pb, 23), LP(pb, 24), LP(pb, 25), LP(pb, 26), yg);
    // SAGE into gene
    run_sage(xb2, xg, ei_rgb, E_rgb, NGg, LP(pb, 15), LP(pb, 16), yg);
    relu_k<<<cdiv(NGg * 64, BK), BK, 0, stream>>>(yg, NGg * 64);

    // msig = gat(genemsig)
    fill_bias_k<<<cdiv(NMm * 64, BK), BK, 0, stream>>>(
        ym, LP(pb, 9), nullptr, nullptr, nullptr, NMm * 64);
    run_gat(xg, xm, ei_gm, E_gm, NGg, NMm, NMm,
            LP(pb, 5), LP(pb, 6), LP(pb, 7), LP(pb, 8), ym);
    relu_k<<<cdiv(NMm * 64, BK), BK, 0, stream>>>(ym, NMm * 64);

    // reactome = gat(genereact)
    fill_bias_k<<<cdiv(NRr * 64, BK), BK, 0, stream>>>(
        yr, LP(pb, 14), nullptr, nullptr, nullptr, NRr * 64);
    run_gat(xg, xr, ei_gr, E_gr, NGg, NRr, NRr,
            LP(pb, 10), LP(pb, 11), LP(pb, 12), LP(pb, 13), yr);
    relu_k<<<cdiv(NRr * 64, BK), BK, 0, stream>>>(yr, NRr * 64);

    // bp = sage(genebp)
    fill_bias_k<<<cdiv(NBb * 64, BK), BK, 0, stream>>>(
        yb, LP(pb, 4), nullptr, nullptr, nullptr, NBb * 64);
    run_sage(xg, xb2, ei_gb, E_gb, NBb, LP(pb, 2), LP(pb, 3), yb);
    relu_k<<<cdiv(NBb * 64, BK), BK, 0, stream>>>(yb, NBb * 64);
  };

  // ---- encoder ----
  run_layer(31, x_gene, x_msig, x_react, x_bp, og, om, orr, ob);   // layer 0 -> ws
  run_layer(59, og, om, orr, ob, zg, zm, zr, zb);                  // layer 1 -> d_out

  // ---- decoders (output order: g2g, genebp, genemsig, genereact) ----
  decode_wmma<<<cdiv(EL_g2g, 128), BK, 0, stream>>>(
      zg, zg, eli_g2g, eli_g2g + EL_g2g, EL_g2g,
      F(19), F(21), F(20), F(22), o_g2g);                 // dec.gene_gene
  decode_wmma<<<cdiv(EL_gb, 128), BK, 0, stream>>>(
      zg, zb, eli_gb, eli_gb + EL_gb, EL_gb,
      F(15), F(17), F(16), F(18), o_gb);                  // dec.gene_bp
  decode_wmma<<<cdiv(EL_gm, 128), BK, 0, stream>>>(
      zg, zm, eli_gm, eli_gm + EL_gm, EL_gm,
      F(23), F(25), F(24), F(26), o_gm);                  // dec.gene_msig
  decode_wmma<<<cdiv(EL_gr, 128), BK, 0, stream>>>(
      zg, zr, eli_gr, eli_gr + EL_gr, EL_gr,
      F(27), F(29), F(28), F(30), o_gr);                  // dec.gene_reactome
}
